// DR2FWL2Kernel_88021059764491
// MI455X (gfx1250) — compile-verified, matrix operand order checked by reading, not run on hardware
//
#include <hip/hip_runtime.h>
#include <cstddef>

// ---------------------------------------------------------------------------
// DR2FWL2 forward for MI455X (gfx1250, wave32, WMMA).
//   - Triangle aggregation: one wave per triangle, float2/lane gathers,
//     global_atomic_add_f32 scatters (f32-exact accumulation).
//   - Fused MLP(64->128->64) + LayerNorm + ReLU + residual per 16-row tile
//     using v_wmma_f32_16x16x32_bf16 (f32 accumulators, bf16 operands).
//   - Final 64x64 projection with the same WMMA tiling.
// ---------------------------------------------------------------------------

typedef __attribute__((ext_vector_type(16))) __bf16 v16bf;
typedef __attribute__((ext_vector_type(8)))  float  v8f;

static __device__ __forceinline__ v8f wmma_bf16(const v16bf& A, const v16bf& B,
                                                const v8f& C) {
  return __builtin_amdgcn_wmma_f32_16x16x32_bf16(
      /*neg_a=*/false, A, /*neg_b=*/false, B,
      /*c_mod=*/(short)0, C, /*reuse_a=*/false, /*reuse_b=*/false);
}

// ---------------------------------------------------------------------------
// Weight pre-swizzle: row-major W[K][N] (f32)  ->  bf16 B-operand layout
//   dst[((kt*NT + nt)*32 + lane)*16 + j]
//   lane<16: N = nt*16 + lane,      K halves j: kt*32 + {0..7, 16..23}
//   lane>=16: N = nt*16 + lane-16,  K halves j: kt*32 + {8..15, 24..31}
// ---------------------------------------------------------------------------
__global__ void prep_w_kernel(const float* __restrict__ W, __bf16* __restrict__ dst,
                              int KT, int NT) {
  int idx = blockIdx.x * blockDim.x + threadIdx.x;
  int total = KT * NT * 512;
  if (idx >= total) return;
  int j    = idx & 15;
  int lane = (idx >> 4) & 31;
  int tile = idx >> 9;
  int nt   = tile % NT;
  int kt   = tile / NT;
  int n    = nt * 16 + (lane & 15);
  int klo  = (lane < 16) ? 0 : 8;
  int k    = kt * 32 + ((j < 8) ? (klo + j) : (16 + klo + (j - 8)));
  dst[idx] = (__bf16)W[(size_t)k * (NT * 16) + n];
}

// ---------------------------------------------------------------------------
// Triangle aggregation. One wave per triangle; lane handles 2 channels.
//   agg[h0][e0] += f[h1][e1]        * f[h2][e2]
//   agg[h1][e1] += f[h0][e0]        * f[h2][inv[h2][e2]]
//   agg[h2][e2] += f[h1][inv[h1][e1]] * f[h0][e0]
// ---------------------------------------------------------------------------
template <int H0, int H1, int H2>
__global__ __launch_bounds__(256)
void tri_agg_kernel(const int* __restrict__ tri, int T,
                    const float* __restrict__ x1, const float* __restrict__ x2,
                    float* __restrict__ a1, float* __restrict__ a2,
                    const int* __restrict__ inv1, const int* __restrict__ inv2) {
  int wv = blockIdx.x * (blockDim.x >> 5) + (threadIdx.x >> 5);
  if (wv >= T) return;
  int lane = threadIdx.x & 31;

  int e0 = tri[wv];
  int e1 = tri[(size_t)T + wv];
  int e2 = tri[2 * (size_t)T + wv];

  const float* f0 = H0 ? x2 : x1;
  const float* f1 = H1 ? x2 : x1;
  const float* f2 = H2 ? x2 : x1;
  float* g0 = H0 ? a2 : a1;
  float* g1 = H1 ? a2 : a1;
  float* g2 = H2 ? a2 : a1;
  const int* i1 = H1 ? inv2 : inv1;
  const int* i2 = H2 ? inv2 : inv1;

  int ie1 = i1[e1];
  int ie2 = i2[e2];

  int c = lane * 2;
  float2 ve1  = *(const float2*)(f1 + (size_t)e1  * 64 + c);
  float2 ve2  = *(const float2*)(f2 + (size_t)e2  * 64 + c);
  float2 ve0  = *(const float2*)(f0 + (size_t)e0  * 64 + c);
  float2 vie2 = *(const float2*)(f2 + (size_t)ie2 * 64 + c);
  float2 vie1 = *(const float2*)(f1 + (size_t)ie1 * 64 + c);

  float* p0 = g0 + (size_t)e0 * 64 + c;
  atomicAdd(p0,     ve1.x * ve2.x);
  atomicAdd(p0 + 1, ve1.y * ve2.y);
  float* p1 = g1 + (size_t)e1 * 64 + c;
  atomicAdd(p1,     ve0.x * vie2.x);
  atomicAdd(p1 + 1, ve0.y * vie2.y);
  float* p2 = g2 + (size_t)e2 * 64 + c;
  atomicAdd(p2,     vie1.x * ve0.x);
  atomicAdd(p2 + 1, vie1.y * ve0.y);
}

// ---------------------------------------------------------------------------
// Fused MLP + LayerNorm + ReLU + residual. One wave = one 16-row tile.
//   hidden = relu(A(16x64) @ Wa(64x128) + ba)      (2 k-tiles x 8 n-tiles)
//   y      = A2(16x128) @ Wb(128x64) + bb          (4 k-tiles x 4 n-tiles)
//   x_out  = relu(LN(y));  s += x_out
// Per-wave 4KB LDS region: hidden staging (bf16 16x128), then reused as
// LN staging (f32 16x64). Same-wave DS ops are in-order -> no barriers.
// ---------------------------------------------------------------------------
__global__ __launch_bounds__(256)
void mlp_ln_kernel(const float* __restrict__ a, float* __restrict__ x,
                   float* __restrict__ s,
                   const __bf16* __restrict__ Wa, const float* __restrict__ ba,
                   const __bf16* __restrict__ Wb, const float* __restrict__ bb,
                   const float* __restrict__ g, const float* __restrict__ be,
                   int E, int isFirstLayer) {
  __shared__ __align__(16) unsigned char sh_buf[8][4096];

  const int lane = threadIdx.x & 31;
  const int wv   = threadIdx.x >> 5;
  const int tile = blockIdx.x * 8 + wv;
  if (tile * 16 >= E) return;  // wave-uniform; no barriers below

  const int rowStart = tile * 16;
  const int mrow = lane & 15;   // M row within tile (A layout); N column in C layout
  const int hi   = lane >> 4;   // half selector
  const int klo  = hi ? 8 : 0;

  // ---- GEMM1 ----
  v8f h[8];
#pragma unroll
  for (int nt = 0; nt < 8; ++nt)
#pragma unroll
    for (int r = 0; r < 8; ++r) h[nt][r] = 0.0f;

#pragma unroll
  for (int kt = 0; kt < 2; ++kt) {
    const float* rp = a + (size_t)(rowStart + mrow) * 64 + kt * 32 + klo;
    v16bf A;
#pragma unroll
    for (int j = 0; j < 8; ++j) {
      A[j]     = (__bf16)rp[j];
      A[8 + j] = (__bf16)rp[16 + j];
    }
#pragma unroll
    for (int nt = 0; nt < 8; ++nt) {
      v16bf B = *(const v16bf*)(Wa + ((size_t)(kt * 8 + nt) * 32 + lane) * 16);
      h[nt] = wmma_bf16(A, B, h[nt]);
    }
  }

  // bias + relu -> bf16 hidden staged in per-wave LDS (row-major 16x128)
  __bf16* hid = reinterpret_cast<__bf16*>(sh_buf[wv]);
#pragma unroll
  for (int nt = 0; nt < 8; ++nt) {
#pragma unroll
    for (int r = 0; r < 8; ++r) {
      int row = r + 8 * hi;
      int col = nt * 16 + mrow;
      float v = h[nt][r] + ba[col];
      hid[row * 128 + col] = (__bf16)(v > 0.0f ? v : 0.0f);
    }
  }

  // ---- GEMM2 ----
  v8f o[4];
#pragma unroll
  for (int nt = 0; nt < 4; ++nt)
#pragma unroll
    for (int r = 0; r < 8; ++r) o[nt][r] = 0.0f;

#pragma unroll
  for (int kt = 0; kt < 4; ++kt) {
    const __bf16* rp = hid + mrow * 128 + kt * 32 + klo;
    v16bf A;
#pragma unroll
    for (int j = 0; j < 8; ++j) {
      A[j]     = rp[j];
      A[8 + j] = rp[16 + j];
    }
#pragma unroll
    for (int nt = 0; nt < 4; ++nt) {
      v16bf B = *(const v16bf*)(Wb + ((size_t)(kt * 4 + nt) * 32 + lane) * 16);
      o[nt] = wmma_bf16(A, B, o[nt]);
    }
  }

  // bias -> f32 LN staging (reuses same per-wave LDS; same-wave DS in-order)
  float* lnb = reinterpret_cast<float*>(sh_buf[wv]);
#pragma unroll
  for (int nt = 0; nt < 4; ++nt) {
#pragma unroll
    for (int r = 0; r < 8; ++r) {
      int row = r + 8 * hi;
      int col = nt * 16 + mrow;
      lnb[row * 64 + col] = o[nt][r] + bb[col];
    }
  }

  // ---- LayerNorm + ReLU + residual: lane pair (l, l^16) shares row mrow ----
  {
    const float* vp = lnb + mrow * 64 + hi * 32;
    float vals[32];
    float sum = 0.0f, sq = 0.0f;
#pragma unroll
    for (int c2 = 0; c2 < 32; ++c2) {
      float v = vp[c2];
      vals[c2] = v;
      sum += v;
      sq += v * v;
    }
    sum += __shfl_xor(sum, 16);
    sq  += __shfl_xor(sq, 16);
    float mu  = sum * (1.0f / 64.0f);
    float var = sq * (1.0f / 64.0f) - mu * mu;
    float inv = rsqrtf(var + 1e-5f);

    size_t base = (size_t)(rowStart + mrow) * 64 + hi * 32;
#pragma unroll
    for (int c2 = 0; c2 < 32; ++c2) {
      int col = hi * 32 + c2;
      float y = (vals[c2] - mu) * inv * g[col] + be[col];
      y = y > 0.0f ? y : 0.0f;
      float sv = isFirstLayer ? 0.0f : s[base + c2];
      s[base + c2] = sv + y;
      x[base + c2] = y;
    }
  }
}

// ---------------------------------------------------------------------------
// Final projection: out = s @ Wf + bf.  16-row tile per wave, 2 k-tiles,
// 4 n-tiles of WMMA.
// ---------------------------------------------------------------------------
__global__ __launch_bounds__(256)
void proj_kernel(const float* __restrict__ s, const __bf16* __restrict__ Wf,
                 const float* __restrict__ bf, float* __restrict__ out, int E) {
  const int lane = threadIdx.x & 31;
  const int wv   = threadIdx.x >> 5;
  const int tile = blockIdx.x * 8 + wv;
  if (tile * 16 >= E) return;

  const int rowStart = tile * 16;
  const int mrow = lane & 15;
  const int hi   = lane >> 4;
  const int klo  = hi ? 8 : 0;

  v8f o[4];
#pragma unroll
  for (int nt = 0; nt < 4; ++nt)
#pragma unroll
    for (int r = 0; r < 8; ++r) o[nt][r] = 0.0f;

#pragma unroll
  for (int kt = 0; kt < 2; ++kt) {
    const float* rp = s + (size_t)(rowStart + mrow) * 64 + kt * 32 + klo;
    v16bf A;
#pragma unroll
    for (int j = 0; j < 8; ++j) {
      A[j]     = (__bf16)rp[j];
      A[8 + j] = (__bf16)rp[16 + j];
    }
#pragma unroll
    for (int nt = 0; nt < 4; ++nt) {
      v16bf B = *(const v16bf*)(Wf + ((size_t)(kt * 4 + nt) * 32 + lane) * 16);
      o[nt] = wmma_bf16(A, B, o[nt]);
    }
  }

#pragma unroll
  for (int nt = 0; nt < 4; ++nt) {
#pragma unroll
    for (int r = 0; r < 8; ++r) {
      int row = r + 8 * hi;
      int col = nt * 16 + mrow;
      out[(size_t)(rowStart + row) * 64 + col] = o[nt][r] + bf[col];
    }
  }
}

// ---------------------------------------------------------------------------
extern "C" void kernel_launch(void* const* d_in, const int* in_sizes, int n_in,
                              void* d_out, int out_size, void* d_ws, size_t ws_size,
                              hipStream_t stream) {
  (void)n_in; (void)out_size; (void)ws_size;

  const float* edge1 = (const float*)d_in[0];
  const float* edge2 = (const float*)d_in[1];
  const int* t111 = (const int*)d_in[2];
  const int* t112 = (const int*)d_in[3];
  const int* t122 = (const int*)d_in[4];
  const int* t222 = (const int*)d_in[5];
  const int* inv1 = (const int*)d_in[6];
  const int* inv2 = (const int*)d_in[7];
  const float* W1a = (const float*)d_in[8];
  const float* b1a = (const float*)d_in[9];
  const float* W1b = (const float*)d_in[10];
  const float* b1b = (const float*)d_in[11];
  const float* W2a = (const float*)d_in[12];
  const float* b2a = (const float*)d_in[13];
  const float* W2b = (const float*)d_in[14];
  const float* b2b = (const float*)d_in[15];
  const float* g1  = (const float*)d_in[16];
  const float* be1 = (const float*)d_in[17];
  const float* g2  = (const float*)d_in[18];
  const float* be2 = (const float*)d_in[19];
  const float* Wf1 = (const float*)d_in[20];
  const float* bf1 = (const float*)d_in[21];
  const float* Wf2 = (const float*)d_in[22];
  const float* bf2 = (const float*)d_in[23];

  const int C = 64, H = 128, L = 3;
  const int E1 = in_sizes[0] / C;
  const int E2 = in_sizes[1] / C;
  const int T111 = in_sizes[2] / 3;
  const int T112 = in_sizes[3] / 3;
  const int T122 = in_sizes[4] / 3;
  const int T222 = in_sizes[5] / 3;

  const size_t E1C = (size_t)E1 * C;
  const size_t E2C = (size_t)E2 * C;

  // workspace layout (f32 region, then bf16 weight region)
  float* wsf = (float*)d_ws;
  float* x1 = wsf;
  float* a1 = x1 + E1C;
  float* s1 = a1 + E1C;
  float* x2 = s1 + E1C;
  float* a2 = x2 + E2C;
  float* s2 = a2 + E2C;
  __bf16* hb  = (__bf16*)(s2 + E2C);
  __bf16* w1a = hb;                 // L * 8192
  __bf16* w1b = hb + 3 * 8192;      // L * 8192
  __bf16* w2a = hb + 6 * 8192;
  __bf16* w2b = hb + 9 * 8192;
  __bf16* wf1 = hb + 12 * 8192;     // 4096
  __bf16* wf2 = wf1 + 4096;

  // ---- pre-swizzle all weights to bf16 B-operand layout ----
  for (int l = 0; l < L; ++l) {
    prep_w_kernel<<<(2 * 8 * 512 + 255) / 256, 256, 0, stream>>>(
        W1a + (size_t)l * C * H, w1a + (size_t)l * 8192, 2, 8);
    prep_w_kernel<<<(4 * 4 * 512 + 255) / 256, 256, 0, stream>>>(
        W1b + (size_t)l * H * C, w1b + (size_t)l * 8192, 4, 4);
    prep_w_kernel<<<(2 * 8 * 512 + 255) / 256, 256, 0, stream>>>(
        W2a + (size_t)l * C * H, w2a + (size_t)l * 8192, 2, 8);
    prep_w_kernel<<<(4 * 4 * 512 + 255) / 256, 256, 0, stream>>>(
        W2b + (size_t)l * H * C, w2b + (size_t)l * 8192, 4, 4);
  }
  prep_w_kernel<<<(2 * 4 * 512 + 255) / 256, 256, 0, stream>>>(Wf1, wf1, 2, 4);
  prep_w_kernel<<<(2 * 4 * 512 + 255) / 256, 256, 0, stream>>>(Wf2, wf2, 2, 4);

  // ---- init feature buffers ----
  hipMemcpyAsync(x1, edge1, E1C * sizeof(float), hipMemcpyDeviceToDevice, stream);
  hipMemcpyAsync(x2, edge2, E2C * sizeof(float), hipMemcpyDeviceToDevice, stream);

  const int mlpGrid1 = (E1 / 16 + 7) / 8;
  const int mlpGrid2 = (E2 / 16 + 7) / 8;

  for (int l = 0; l < L; ++l) {
    // a = x  (EPS = EPS2 = 0), then triangle scatter-adds on top
    hipMemcpyAsync(a1, x1, E1C * sizeof(float), hipMemcpyDeviceToDevice, stream);
    hipMemcpyAsync(a2, x2, E2C * sizeof(float), hipMemcpyDeviceToDevice, stream);

    tri_agg_kernel<0, 0, 0><<<(T111 + 7) / 8, 256, 0, stream>>>(
        t111, T111, x1, x2, a1, a2, inv1, inv2);
    tri_agg_kernel<0, 0, 1><<<(T112 + 7) / 8, 256, 0, stream>>>(
        t112, T112, x1, x2, a1, a2, inv1, inv2);
    tri_agg_kernel<0, 1, 1><<<(T122 + 7) / 8, 256, 0, stream>>>(
        t122, T122, x1, x2, a1, a2, inv1, inv2);
    tri_agg_kernel<1, 1, 1><<<(T222 + 7) / 8, 256, 0, stream>>>(
        t222, T222, x1, x2, a1, a2, inv1, inv2);

    mlp_ln_kernel<<<mlpGrid1, 256, 0, stream>>>(
        a1, x1, s1,
        w1a + (size_t)l * 8192, b1a + (size_t)l * H,
        w1b + (size_t)l * 8192, b1b + (size_t)l * C,
        g1 + (size_t)l * C, be1 + (size_t)l * C, E1, l == 0);
    mlp_ln_kernel<<<mlpGrid2, 256, 0, stream>>>(
        a2, x2, s2,
        w2a + (size_t)l * 8192, b2a + (size_t)l * H,
        w2b + (size_t)l * 8192, b2b + (size_t)l * C,
        g2 + (size_t)l * C, be2 + (size_t)l * C, E2, l == 0);
  }

  // ---- final projection into d_out (out1 then out2, flat) ----
  float* out = (float*)d_out;
  proj_kernel<<<mlpGrid1, 256, 0, stream>>>(s1, wf1, bf1, out, E1);
  proj_kernel<<<mlpGrid2, 256, 0, stream>>>(s2, wf2, bf2, out + E1C, E2);
}
